// GATEncoder_40767829574199
// MI455X (gfx1250) — matrix operand order repaired and checked by reading, not compiled, not run on hardware
//
#include <hip/hip_runtime.h>
#include <math.h>

#define NN 100000
#define NE 1600000

typedef __attribute__((ext_vector_type(16))) _Float16 v16h;
typedef __attribute__((ext_vector_type(8)))  float    v8f;

// ---- monotonic float<->uint encoding for atomic segment-max (memset-0 == -inf) ----
__device__ __forceinline__ unsigned fenc(float f) {
  unsigned u = __float_as_uint(f);
  return (u & 0x80000000u) ? ~u : (u | 0x80000000u);
}
__device__ __forceinline__ float fdec(unsigned u) {
  unsigned v = (u & 0x80000000u) ? (u & 0x7fffffffu) : ~u;
  return __uint_as_float(v);
}

// ---------------- WMMA GEMM: C[M,Nout] = act(A[M,K] @ W[Nout,K]^T + bias) ----------------
// grid = (ceil(M/16), Nout/16), block = 32 (one wave per 16x16 tile). K <= 64.
__global__ void k_gemm_wmma(const float* __restrict__ A, const float* __restrict__ W,
                            const float* __restrict__ bias, float* __restrict__ C,
                            int M, int K, int Nout, int act) {
  __shared__ _Float16 sA[16][64];
  __shared__ _Float16 sB[16][64];
  const int lane = threadIdx.x;
  const int row0 = blockIdx.x * 16;
  const int col0 = blockIdx.y * 16;
  for (int i = lane; i < 16 * 64; i += 32) {
    int r = i >> 6, k = i & 63;
    float va = 0.f, vb = 0.f;
    if (k < K) {
      int ar = row0 + r;
      if (ar < M) va = A[(long)ar * K + k];
      vb = W[(long)(col0 + r) * K + k];
    }
    sA[r][k] = (_Float16)va;
    sB[r][k] = (_Float16)vb;
  }
  __syncthreads();
  const int mrow = lane & 15;          // A row / B column for this lane
  const int kb   = (lane >> 4) * 8;    // A K-base per lane half (ISA layout)
  const int kcb  = (lane >> 4) * 16;   // unused; B loads also use kb
  v8f acc = {0.f, 0.f, 0.f, 0.f, 0.f, 0.f, 0.f, 0.f};
  const int ktiles = (K + 31) >> 5;
  for (int kt = 0; kt < ktiles; ++kt) {
    int k0 = kt * 32;
    v16h a, b;
#pragma unroll
    for (int h = 0; h < 16; ++h) {
      a[h] = sA[mrow][k0 + kb + h + (h >= 8 ? 8 : 0)];  // K = 0-7,16-23 (lo) / 8-15,24-31 (hi)
      b[h] = sB[mrow][k0 + kb + h + (h >= 8 ? 8 : 0)];
    }
    acc = __builtin_amdgcn_wmma_f32_16x16x32_f16(false, a, false, b, (short)0, acc, false, false);
  }
  const int ncol = lane & 15;
  const int mb   = (lane >> 4) * 8;    // C/D: VGPR v -> M = v + 8*(lane>=16)
#pragma unroll
  for (int v = 0; v < 8; ++v) {
    int r = row0 + mb + v;
    if (r < M) {
      float val = acc[v] + bias[col0 + ncol];
      if (act == 1) val = fmaxf(val, 0.f);
      else if (act == 2) val = (val > 0.f) ? val : (__expf(val) - 1.f);
      C[(long)r * Nout + col0 + ncol] = val;
    }
  }
}

// ---------------- small helper kernels ----------------
__global__ void k_concat(const float* __restrict__ x, const int* __restrict__ nt,
                         const float* __restrict__ emb, float* __restrict__ xin, int n) {
  long i = (long)blockIdx.x * blockDim.x + threadIdx.x;
  if (i >= (long)n * 24) return;
  int r = (int)(i / 24), c = (int)(i - (long)r * 24);
  xin[i] = (c < 16) ? x[(long)r * 16 + c] : emb[nt[r] * 8 + (c - 16)];
}

__global__ void k_degloop(const int* __restrict__ dst, const float* __restrict__ ea,
                          float* __restrict__ deg, float* __restrict__ loopsum, int e) {
  long i = (long)blockIdx.x * blockDim.x + threadIdx.x;
  if (i >= e) return;
  int d = dst[i];
  atomicAdd(&deg[d], 1.f);
#pragma unroll
  for (int j = 0; j < 8; ++j) atomicAdd(&loopsum[(long)d * 8 + j], ea[i * 8 + j]);
}

__global__ void k_loopdiv(float* __restrict__ loopat, const float* __restrict__ deg, int n) {
  long i = (long)blockIdx.x * blockDim.x + threadIdx.x;
  if (i >= (long)n * 8) return;
  loopat[i] /= fmaxf(deg[i >> 3], 1.f);
}

// ---------------- edge logits + segment max (one wave per edge; We folded in from LDS) ----
__global__ void k_logit(const int* __restrict__ src, const int* __restrict__ dst,
                        const float* __restrict__ ea, const float* __restrict__ loopat,
                        const float* __restrict__ xl, const float* __restrict__ xr,
                        const float* __restrict__ We, const float* __restrict__ att,
                        float* __restrict__ logit, unsigned* __restrict__ maxenc, int E, int n) {
  __shared__ float sWe[128 * 8];
  __shared__ float sAtt[128];
  for (int i = threadIdx.x; i < 1024; i += blockDim.x) sWe[i] = We[i];
  for (int i = threadIdx.x; i < 128; i += blockDim.x) sAtt[i] = att[i];
  __syncthreads();
  const int lane = threadIdx.x & 31;
  long t = (long)blockIdx.x * (blockDim.x >> 5) + (threadIdx.x >> 5);
  long ET = (long)E + n;
  if (t >= ET) return;
  int s, d; const float* eap;
  if (t < E) { s = src[t]; d = dst[t]; eap = ea + t * 8; }
  else       { s = d = (int)(t - E);   eap = loopat + (long)s * 8; }
  float eav[8];
#pragma unroll
  for (int j = 0; j < 8; ++j) eav[j] = eap[j];
  float part = 0.f;
#pragma unroll
  for (int u = 0; u < 4; ++u) {
    int c = lane * 4 + u;                     // c<64: head0, else head1
    float ew = 0.f;
#pragma unroll
    for (int j = 0; j < 8; ++j) ew = fmaf(eav[j], sWe[c * 8 + j], ew);
    float v = xl[(long)s * 128 + c] + xr[(long)d * 128 + c] + ew;
    v = (v > 0.f) ? v : 0.2f * v;             // leaky_relu(0.2)
    part += v * sAtt[c];
  }
  part += __shfl_xor(part, 8, 16);
  part += __shfl_xor(part, 4, 16);
  part += __shfl_xor(part, 2, 16);
  part += __shfl_xor(part, 1, 16);
  if ((lane & 15) == 0) {
    int hI = lane >> 4;
    logit[t * 2 + hI] = part;
    atomicMax(&maxenc[(long)d * 2 + hI], fenc(part));
  }
}

__global__ void k_expdenom(const int* __restrict__ dst, float* __restrict__ logit,
                           const unsigned* __restrict__ maxenc, float* __restrict__ denom,
                           int E, int n) {
  long i = (long)blockIdx.x * blockDim.x + threadIdx.x;
  long ET2 = 2L * ((long)E + n);
  if (i >= ET2) return;
  long t = i >> 1; int hI = (int)(i & 1);
  int d = (t < E) ? dst[t] : (int)(t - E);
  float a = __expf(logit[i] - fdec(maxenc[(long)d * 2 + hI]));
  logit[i] = a;                                // store numerator in place
  atomicAdd(&denom[(long)d * 2 + hI], a);
}

__global__ void k_aggregate(const int* __restrict__ src, const int* __restrict__ dst,
                            const float* __restrict__ logit, const float* __restrict__ denom,
                            const float* __restrict__ xl, float* __restrict__ accum, int E, int n) {
  const int lane = threadIdx.x & 31;
  long t = (long)blockIdx.x * (blockDim.x >> 5) + (threadIdx.x >> 5);
  long ET = (long)E + n;
  if (t >= ET) return;
  int s, d;
  if (t < E) { s = src[t]; d = dst[t]; } else { s = d = (int)(t - E); }
  int hI = lane >> 4;
  float coeff = logit[t * 2 + hI] / denom[(long)d * 2 + hI];
#pragma unroll
  for (int u = 0; u < 4; ++u) {
    int c = lane * 4 + u;
    atomicAdd(&accum[(long)d * 128 + c], coeff * xl[(long)s * 128 + c]);
  }
}

__global__ void k_finalize(const float* __restrict__ accum, const float* __restrict__ bias,
                           float* __restrict__ out, int n, int act) {
  long i = (long)blockIdx.x * blockDim.x + threadIdx.x;
  if (i >= (long)n * 64) return;
  int r = (int)(i >> 6), c = (int)(i & 63);
  float v = 0.5f * (accum[(long)r * 128 + c] + accum[(long)r * 128 + 64 + c]) + bias[c];
  if (act == 2) v = (v > 0.f) ? v : (__expf(v) - 1.f);
  out[i] = v;
}

__global__ void k_colsum(const float* __restrict__ h, float* __restrict__ colsum, int n) {
  __shared__ float sc[256];
  int c = threadIdx.x & 63;
  int rg = threadIdx.x >> 6;
  int r0 = blockIdx.x * 256;
  int r1 = (r0 + 256 < n) ? (r0 + 256) : n;
  float acc = 0.f;
  for (int r = r0 + rg; r < r1; r += 4) acc += h[(long)r * 64 + c];
  sc[threadIdx.x] = acc;
  __syncthreads();
  if (rg == 0) atomicAdd(&colsum[c], sc[c] + sc[64 + c] + sc[128 + c] + sc[192 + c]);
}

__global__ void k_gvec(const float* __restrict__ colsum, const float* __restrict__ Wg,
                       const float* __restrict__ bg, float* __restrict__ g, int n) {
  int o = threadIdx.x;
  if (o >= 64) return;
  float acc = bg[o];
  float inv = 1.f / (float)n;
#pragma unroll
  for (int k = 0; k < 64; ++k) acc = fmaf(colsum[k] * inv, Wg[o * 64 + k], acc);
  g[o] = tanhf(acc);
}

// ---------------- host launch ----------------
extern "C" void kernel_launch(void* const* d_in, const int* in_sizes, int n_in,
                              void* d_out, int out_size, void* d_ws, size_t ws_size,
                              hipStream_t stream) {
  const float* x    = (const float*)d_in[0];
  const int*  ntype = (const int*)  d_in[1];
  const int*  eidx  = (const int*)  d_in[2];
  const float* ea   = (const float*)d_in[3];
  const float* emb  = (const float*)d_in[4];
  const float* W_in = (const float*)d_in[5];
  const float* b_in = (const float*)d_in[6];
  const float* Wl1  = (const float*)d_in[7];
  const float* bl1  = (const float*)d_in[8];
  const float* Wr1  = (const float*)d_in[9];
  const float* br1  = (const float*)d_in[10];
  const float* We1  = (const float*)d_in[11];
  const float* att1 = (const float*)d_in[12];
  const float* bias1= (const float*)d_in[13];
  const float* Wl2  = (const float*)d_in[14];
  const float* bl2  = (const float*)d_in[15];
  const float* Wr2  = (const float*)d_in[16];
  const float* br2  = (const float*)d_in[17];
  const float* We2  = (const float*)d_in[18];
  const float* att2 = (const float*)d_in[19];
  const float* bias2= (const float*)d_in[20];
  const float* Wg   = (const float*)d_in[21];
  const float* bg   = (const float*)d_in[22];

  const int n = NN, e = NE;
  const int* srcp = eidx;
  const int* dstp = eidx + e;
  const long ET = (long)e + n;

  float* ws = (float*)d_ws;
  size_t o = 0;
  float* xin    = ws + o; o += (size_t)n * 24;
  float* h0     = ws + o; o += (size_t)n * 64;
  float* h1     = ws + o; o += (size_t)n * 64;
  float* xl     = ws + o; o += (size_t)n * 128;
  float* xr     = ws + o; o += (size_t)n * 128;
  float* loopat = ws + o; o += (size_t)n * 8;
  float* deg    = ws + o; o += (size_t)n;
  float* logit  = ws + o; o += (size_t)(2L * ET);
  unsigned* maxenc = (unsigned*)(ws + o); o += (size_t)(2L * n);
  float* denom  = ws + o; o += (size_t)(2L * n);
  float* accum  = ws + o; o += (size_t)n * 128;
  float* colsum = ws + o; o += 64;

  float* out_h = (float*)d_out;
  float* out_g = out_h + (size_t)n * 64;

  auto cdiv = [](long a, long b) { return (int)((a + b - 1) / b); };

  hipMemsetAsync(deg, 0, (size_t)n * 4, stream);
  hipMemsetAsync(loopat, 0, (size_t)n * 8 * 4, stream);
  hipMemsetAsync(colsum, 0, 64 * 4, stream);

  k_concat<<<cdiv((long)n * 24, 256), 256, 0, stream>>>(x, ntype, emb, xin, n);
  k_gemm_wmma<<<dim3(cdiv(n, 16), 4), 32, 0, stream>>>(xin, W_in, b_in, h0, n, 24, 64, 1);
  k_degloop<<<cdiv(e, 256), 256, 0, stream>>>(dstp, ea, deg, loopat, e);
  k_loopdiv<<<cdiv((long)n * 8, 256), 256, 0, stream>>>(loopat, deg, n);

  // ---- GATv2 layer 1 (ELU out) ----
  hipMemsetAsync(maxenc, 0, (size_t)2 * n * 4, stream);
  hipMemsetAsync(denom, 0, (size_t)2 * n * 4, stream);
  hipMemsetAsync(accum, 0, (size_t)n * 128 * 4, stream);
  k_gemm_wmma<<<dim3(cdiv(n, 16), 8), 32, 0, stream>>>(h0, Wl1, bl1, xl, n, 64, 128, 0);
  k_gemm_wmma<<<dim3(cdiv(n, 16), 8), 32, 0, stream>>>(h0, Wr1, br1, xr, n, 64, 128, 0);
  k_logit<<<cdiv(ET, 8), 256, 0, stream>>>(srcp, dstp, ea, loopat, xl, xr, We1, att1, logit, maxenc, e, n);
  k_expdenom<<<cdiv(ET * 2, 256), 256, 0, stream>>>(dstp, logit, maxenc, denom, e, n);
  k_aggregate<<<cdiv(ET, 8), 256, 0, stream>>>(srcp, dstp, logit, denom, xl, accum, e, n);
  k_finalize<<<cdiv((long)n * 64, 256), 256, 0, stream>>>(accum, bias1, h1, n, 2);

  // ---- GATv2 layer 2 (linear out -> d_out) ----
  hipMemsetAsync(maxenc, 0, (size_t)2 * n * 4, stream);
  hipMemsetAsync(denom, 0, (size_t)2 * n * 4, stream);
  hipMemsetAsync(accum, 0, (size_t)n * 128 * 4, stream);
  k_gemm_wmma<<<dim3(cdiv(n, 16), 8), 32, 0, stream>>>(h1, Wl2, bl2, xl, n, 64, 128, 0);
  k_gemm_wmma<<<dim3(cdiv(n, 16), 8), 32, 0, stream>>>(h1, Wr2, br2, xr, n, 64, 128, 0);
  k_logit<<<cdiv(ET, 8), 256, 0, stream>>>(srcp, dstp, ea, loopat, xl, xr, We2, att2, logit, maxenc, e, n);
  k_expdenom<<<cdiv(ET * 2, 256), 256, 0, stream>>>(dstp, logit, maxenc, denom, e, n);
  k_aggregate<<<cdiv(ET, 8), 256, 0, stream>>>(srcp, dstp, logit, denom, xl, accum, e, n);
  k_finalize<<<cdiv((long)n * 64, 256), 256, 0, stream>>>(accum, bias2, out_h, n, 0);

  // ---- graph readout: g = tanh(mean(h) @ Wg^T + bg) ----
  k_colsum<<<cdiv(n, 256), 256, 0, stream>>>(out_h, colsum, n);
  k_gvec<<<1, 64, 0, stream>>>(colsum, Wg, bg, out_g, n);

  (void)in_sizes; (void)n_in; (void)out_size; (void)ws_size;
}
